// GATv2model_22548578304046
// MI455X (gfx1250) — matrix operand order repaired and checked by reading, not compiled, not run on hardware
//
#include <hip/hip_runtime.h>

// ---------------------------------------------------------------------------
// GATv2 x4 on MI455X (gfx1250). Edge phase is L2-resident (working set
// ~100MB << 192MB L2); projections use v_wmma_f32_16x16x32_f16 with weights
// staged into LDS by the Tensor Data Mover (TENSOR_LOAD_TO_LDS).
// ---------------------------------------------------------------------------

typedef __attribute__((ext_vector_type(16))) _Float16 v16h;
typedef __attribute__((ext_vector_type(8)))  float    v8f;
typedef __attribute__((ext_vector_type(4)))  unsigned u32x4;
typedef __attribute__((ext_vector_type(8)))  unsigned u32x8;

// --- Tensor Data Mover: 2D f32 tile load (global -> LDS), whole matrix -----
// D# per CDNA5 ISA:
//   group0: count=1 | lds_addr | global_addr[56:0] | type=2
//   group1: workgroup_mask=0, data_size=4B, tensor_dim0=dim0, tensor_dim1=dim1,
//           tile_dim0=dim0, tile_dim1=dim1, tensor_dim0_stride=dim0
__device__ __forceinline__ void tdm_load_2d_f32(const float* gptr,
                                                unsigned lds_byte_off,
                                                int dim0, int dim1) {
  unsigned long long ga = (unsigned long long)(size_t)gptr;
  u32x4 g0;
  g0[0] = 1u;                                            // count=1, user D#
  g0[1] = lds_byte_off;                                  // lds_addr
  g0[2] = (unsigned)ga;                                  // global_addr[31:0]
  g0[3] = (unsigned)((ga >> 32) & 0x1FFFFFFu) | (2u << 30); // addr[56:32]|type=2
  u32x8 g1;
  g1[0] = 2u << 16;                                      // data_size = 4 bytes
  g1[1] = ((unsigned)dim0 & 0xFFFFu) << 16;              // tensor_dim0[15:0]
  g1[2] = (((unsigned)dim0 >> 16) & 0xFFFFu) | (((unsigned)dim1 & 0xFFFFu) << 16);
  g1[3] = (((unsigned)dim1 >> 16) & 0xFFFFu) | (((unsigned)dim0 & 0xFFFFu) << 16); // tile_dim0
  g1[4] = (unsigned)dim1 & 0xFFFFu;                      // tile_dim1 (tile_dim2=0)
  g1[5] = (unsigned)dim0;                                // tensor_dim0_stride[31:0]
  g1[6] = 0u;
  g1[7] = 0u;
  asm volatile("tensor_load_to_lds %0, %1" :: "s"(g0), "s"(g1) : "memory");
}

// --- WMMA projection: XL = X @ Wl, XR = X @ Wr -----------------------------
// One wave computes a 16-row strip; loops over 16-col tiles and K/32 tiles.
// A fragment (16x32 f16): lane L holds row M=L&15; khalf=(L<16?0:8);
//   element e -> K = khalf + (e<8 ? e : e+8).
// B fragment (32x16 f16): lane L holds col N=L&15; khalf=(L<16?0:8);
//   element e -> K = khalf + (e<8 ? e : e+8), same as A.
// C (16x16 f32): lane L holds col N=L&15; VGPR r -> row M = r + (L<16?0:8).
template<int FIN, int FOUT>
__global__ __launch_bounds__(256) void proj_wmma(
    const float* __restrict__ X,
    const float* __restrict__ Wl, const float* __restrict__ Wr,
    float* __restrict__ XL, float* __restrict__ XR, int nrows)
{
  __shared__ float sWl[FIN * FOUT];
  __shared__ float sWr[FIN * FOUT];

  if (threadIdx.x < 32) {   // wave 0 issues the DMAs (uniform branch)
    tdm_load_2d_f32(Wl, (unsigned)(size_t)&sWl[0], FOUT, FIN);
    tdm_load_2d_f32(Wr, (unsigned)(size_t)&sWr[0], FOUT, FIN);
    __builtin_amdgcn_s_wait_tensorcnt(0);
  }
  __syncthreads();

  const int lane    = threadIdx.x & 31;
  const int wid     = threadIdx.x >> 5;
  const int rowBase = blockIdx.x * 128 + wid * 16;
  const int mrow    = lane & 15;
  const int khalf   = (lane < 16) ? 0 : 8;
  const int kb      = (lane < 16) ? 0 : 16;
  const int col0    = lane & 15;

  constexpr int KT = FIN / 32;
  constexpr int CT = FOUT / 16;

  // Load A fragments once (row clamped; garbage rows only affect discarded C rows)
  int r = rowBase + mrow; if (r >= nrows) r = nrows - 1;
  v16h a[KT];
#pragma unroll
  for (int kt = 0; kt < KT; ++kt) {
    const float* xp = X + (size_t)r * FIN + kt * 32 + khalf;
#pragma unroll
    for (int e = 0; e < 16; ++e) {
      int k = (e < 8) ? e : (e + 8);
      a[kt][e] = (_Float16)xp[k];
    }
  }

#pragma unroll
  for (int ct = 0; ct < CT; ++ct) {
    const int colIdx = ct * 16 + col0;
    v8f cl = {}; v8f cr = {};
#pragma unroll
    for (int kt = 0; kt < KT; ++kt) {
      v16h bl, br;
#pragma unroll
      for (int e = 0; e < 16; ++e) {
        int k = kt * 32 + khalf + ((e < 8) ? e : (e + 8));
        bl[e] = (_Float16)sWl[k * FOUT + colIdx];
        br[e] = (_Float16)sWr[k * FOUT + colIdx];
      }
      cl = __builtin_amdgcn_wmma_f32_16x16x32_f16(false, a[kt], false, bl,
                                                  (short)0, cl, false, false);
      cr = __builtin_amdgcn_wmma_f32_16x16x32_f16(false, a[kt], false, br,
                                                  (short)0, cr, false, false);
    }
#pragma unroll
    for (int r8 = 0; r8 < 8; ++r8) {
      int row = rowBase + r8 + ((lane < 16) ? 0 : 8);
      if (row < nrows) {
        XL[(size_t)row * FOUT + colIdx] = cl[r8];
        XR[(size_t)row * FOUT + colIdx] = cr[r8];
      }
    }
  }
}

// --- helpers ---------------------------------------------------------------
__device__ __forceinline__ void atomicMaxF(float* addr, float v) {
  if (v >= 0.f) atomicMax((int*)addr, __float_as_int(v));
  else          atomicMin((unsigned int*)addr, __float_as_uint(v));
}

__global__ void fill_f32(float* __restrict__ p, float v, int n) {
  int i = blockIdx.x * blockDim.x + threadIdx.x;
  if (i < n) p[i] = v;
}

// --- edge phase ------------------------------------------------------------
__global__ __launch_bounds__(256) void edge_score(
    const float* __restrict__ XL, const float* __restrict__ XR,
    const float* __restrict__ att,
    const long long* __restrict__ src, const long long* __restrict__ dst,
    float* __restrict__ score, float* __restrict__ mx, int F, int nedges)
{
  int e = blockIdx.x * 8 + (threadIdx.x >> 5);
  if (e >= nedges) return;
  int lane = threadIdx.x & 31;
  int s = (int)src[e], d = (int)dst[e];
  const float* pl = XL + (size_t)s * F;
  const float* pr = XR + (size_t)d * F;
  float acc = 0.f;
  for (int f = lane; f < F; f += 32) {
    float v = pl[f] + pr[f];
    v = (v > 0.f) ? v : 0.2f * v;            // leaky_relu, slope 0.2
    acc += att[f] * v;
  }
  for (int off = 16; off > 0; off >>= 1) acc += __shfl_xor(acc, off, 32);
  if (lane == 0) { score[e] = acc; atomicMaxF(&mx[d], acc); }
}

__global__ __launch_bounds__(256) void edge_exp(
    const long long* __restrict__ dst, float* __restrict__ score,
    const float* __restrict__ mx, float* __restrict__ dn, int nedges)
{
  int e = blockIdx.x * blockDim.x + threadIdx.x;
  if (e >= nedges) return;
  int d = (int)dst[e];
  float ex = __expf(score[e] - mx[d]);
  score[e] = ex;                              // overwrite score with exp value
  atomicAdd(&dn[d], ex);
}

__global__ __launch_bounds__(256) void edge_aggregate(
    const float* __restrict__ XL, const float* __restrict__ ex,
    const float* __restrict__ dn,
    const long long* __restrict__ src, const long long* __restrict__ dst,
    float* __restrict__ out, int F, int nedges)
{
  int e = blockIdx.x * 8 + (threadIdx.x >> 5);
  if (e >= nedges) return;
  int lane = threadIdx.x & 31;
  int s = (int)src[e], d = (int)dst[e];
  float w = ex[e] / (dn[d] + 1e-16f);
  const float* pl = XL + (size_t)s * F;
  float*       po = out + (size_t)d * F;
  for (int f = lane; f < F; f += 32)
    atomicAdd(&po[f], pl[f] * w);
}

__global__ void elu_inplace(float* __restrict__ h, float alpha, int n) {
  int i = blockIdx.x * blockDim.x + threadIdx.x;
  if (i >= n) return;
  float v = h[i];
  h[i] = (v > 0.f) ? v : alpha * expm1f(v);
}

// --- driver ----------------------------------------------------------------
extern "C" void kernel_launch(void* const* d_in, const int* in_sizes, int n_in,
                              void* d_out, int out_size, void* d_ws, size_t ws_size,
                              hipStream_t stream) {
  const float*     x   = (const float*)d_in[0];
  const long long* ei  = (const long long*)d_in[1];   // int64 per reference
  const float* Wl[4]  = {(const float*)d_in[2], (const float*)d_in[5],
                         (const float*)d_in[8], (const float*)d_in[11]};
  const float* Wr[4]  = {(const float*)d_in[3], (const float*)d_in[6],
                         (const float*)d_in[9], (const float*)d_in[12]};
  const float* att[4] = {(const float*)d_in[4], (const float*)d_in[7],
                         (const float*)d_in[10], (const float*)d_in[13]};

  const int Nn = in_sizes[0] / 128;      // 100000
  const int Ee = in_sizes[1] / 2;        // 1600000
  const long long* srcI = ei;
  const long long* dstI = ei + Ee;

  float* ws = (float*)d_ws;
  float* xl = ws;                        // N*64
  float* xr = xl + (size_t)Nn * 64;      // N*64
  float* h  = xr + (size_t)Nn * 64;      // N*64 (layer outputs 1..3)
  float* sc = h  + (size_t)Nn * 64;      // E (score, then exp)
  float* mx = sc + (size_t)Ee;           // N
  float* dn = mx + (size_t)Nn;           // N
  float* outF = (float*)d_out;           // N*32, final layer target

  const int pgrid = (Nn + 127) / 128;
  const int ngrid = (Nn + 255) / 256;
  const int egrid8 = (Ee + 7) / 8;
  const int egrid  = (Ee + 255) / 256;

  for (int L = 0; L < 4; ++L) {
    const float* in   = (L == 0) ? x : h;
    const int    fout = (L == 3) ? 32 : 64;
    float*       tgt  = (L == 3) ? outF : h;
    const float  alpha = (L == 3) ? 32.0f : 1.0f;   // nn.ELU(out_channels) quirk

    // 1) projections via WMMA (TDM-staged weights)
    if (L == 0)
      proj_wmma<128, 64><<<pgrid, 256, 0, stream>>>(in, Wl[L], Wr[L], xl, xr, Nn);
    else if (L == 3)
      proj_wmma<64, 32><<<pgrid, 256, 0, stream>>>(in, Wl[L], Wr[L], xl, xr, Nn);
    else
      proj_wmma<64, 64><<<pgrid, 256, 0, stream>>>(in, Wl[L], Wr[L], xl, xr, Nn);

    // 2) init segment buffers; zero the aggregation target (h is dead after proj)
    fill_f32<<<ngrid, 256, 0, stream>>>(mx, -__builtin_inff(), Nn);
    fill_f32<<<ngrid, 256, 0, stream>>>(dn, 0.0f, Nn);
    fill_f32<<<(Nn * fout + 255) / 256, 256, 0, stream>>>(tgt, 0.0f, Nn * fout);

    // 3) attention scores + segment max
    edge_score<<<egrid8, 256, 0, stream>>>(xl, xr, att[L], srcI, dstI, sc, mx, fout, Ee);
    // 4) exp + segment sum
    edge_exp<<<egrid, 256, 0, stream>>>(dstI, sc, mx, dn, Ee);
    // 5) weighted scatter-add (L2-resident atomics)
    edge_aggregate<<<egrid8, 256, 0, stream>>>(xl, sc, dn, srcI, dstI, tgt, fout, Ee);
    // 6) ELU (alpha=1 inner layers, alpha=32 final)
    elu_inplace<<<(Nn * fout + 255) / 256, 256, 0, stream>>>(tgt, alpha, Nn * fout);
  }
}
